// SAGE_5179730559447
// MI455X (gfx1250) — matrix operand order repaired and checked by reading, not compiled, not run on hardware
//
#include <hip/hip_runtime.h>
#include <math.h>

// ---------------------------------------------------------------------------
// CDNA5 (gfx1250, wave32) two-hop GraphSAGE forward.
// GEMMs run on v_wmma_f32_16x16x32_bf16 with an fp32 -> bf16(hi)+bf16(lo)
// split (3 WMMAs per K-step) to retain ~fp32 precision. Edge aggregation is
// HBM-bandwidth-bound and uses global_atomic_add_f32 scatter-mean.
// ---------------------------------------------------------------------------

typedef unsigned short u16;
typedef __attribute__((ext_vector_type(16))) __bf16 v16bf;
typedef __attribute__((ext_vector_type(8)))  float  v8f;

union BfTile { v16bf v; uint4 q[2]; };

__device__ __forceinline__ u16 f2bf(float f) {
  unsigned u = __float_as_uint(f);
  u += 0x7fffu + ((u >> 16) & 1u);   // round-to-nearest-even
  return (u16)(u >> 16);
}
__device__ __forceinline__ float bf2f(u16 h) {
  return __uint_as_float(((unsigned)h) << 16);
}
__device__ __forceinline__ unsigned hash32(unsigned x) {
  x ^= x >> 16; x *= 0x7feb352dU;
  x ^= x >> 15; x *= 0x846ca68bU;
  x ^= x >> 16; return x;
}

// ------------------------------ utility kernels ----------------------------

__global__ void fz(float* __restrict__ p, long long n) {
  long long i = (long long)blockIdx.x * blockDim.x + threadIdx.x;
  long long st = (long long)gridDim.x * blockDim.x;
  for (; i < n; i += st) p[i] = 0.0f;
}

// One block per edge: scatter-add features (mean numerator) + degree count (denominator).
__global__ void scatter_mean_accum(const int* __restrict__ src,
                                   const int* __restrict__ dst,
                                   const float* __restrict__ feat,
                                   float* __restrict__ acc,
                                   float* __restrict__ cnt, int F) {
  const int e = blockIdx.x;
  const int s = src[e], d = dst[e];
  const float* xs = feat + (size_t)s * F;
  float* ad = acc + (size_t)d * F;
  for (int f = threadIdx.x; f < F; f += blockDim.x) atomicAdd(ad + f, xs[f]);
  if (threadIdx.x == 0) atomicAdd(cnt + d, 1.0f);
}

// agg/max(cnt,1) -> bf16 hi + bf16 lo residual
__global__ void pack_mean_hilo(const float* __restrict__ acc,
                               const float* __restrict__ cnt,
                               u16* __restrict__ hi, u16* __restrict__ lo,
                               int rows, int cols) {
  long long n = (long long)rows * cols;
  long long i = (long long)blockIdx.x * blockDim.x + threadIdx.x;
  long long st = (long long)gridDim.x * blockDim.x;
  for (; i < n; i += st) {
    int row = (int)(i / cols);
    float v = acc[i] / fmaxf(cnt[row], 1.0f);
    u16 h = f2bf(v);
    hi[i] = h;
    lo[i] = f2bf(v - bf2f(h));
  }
}

__global__ void pack_hilo(const float* __restrict__ s, u16* __restrict__ hi,
                          u16* __restrict__ lo, long long n) {
  long long i = (long long)blockIdx.x * blockDim.x + threadIdx.x;
  long long st = (long long)gridDim.x * blockDim.x;
  for (; i < n; i += st) {
    float v = s[i];
    u16 h = f2bf(v);
    hi[i] = h;
    lo[i] = f2bf(v - bf2f(h));
  }
}

// W [K x N] (row-major) -> Wt [Npad x K] bf16 hi/lo, zero-padded cols >= N.
__global__ void pack_wT_hilo(const float* __restrict__ W, u16* __restrict__ hi,
                             u16* __restrict__ lo, int K, int N, int Npad) {
  long long n = (long long)Npad * K;
  long long i = (long long)blockIdx.x * blockDim.x + threadIdx.x;
  long long st = (long long)gridDim.x * blockDim.x;
  for (; i < n; i += st) {
    int nn = (int)(i / K);
    int kk = (int)(i % K);
    float v = (nn < N) ? W[(size_t)kk * N + nn] : 0.0f;
    u16 h = f2bf(v);
    hi[i] = h;
    lo[i] = f2bf(v - bf2f(h));
  }
}

// ------------------------------ WMMA GEMM ----------------------------------
// One wave -> one 16x16 f32 tile of  A1@B1 + A2@B2.
// A*: row-major M x K bf16 (hi/lo).  B*: row-major Npad x K bf16 (hi/lo),
// i.e. W transposed so each lane's K-values come in two contiguous 8-runs.
// ISA VGPR layouts (CDNA5 ISA):
//   A 16x32 bf16:  row = lane&15;  K in {akb..akb+7, 16+akb..16+akb+7},
//                  akb = (lane>=16)*8        -> two 16B loads
//   B 32x16 bf16:  col = lane&15;  K in {bkb..bkb+7, 16+bkb..16+bkb+7},
//                  bkb = (lane>=16)*8 -> two 16B loads from transposed W
//   C/D 16x16 f32: col = lane&15;  row = v + (lane>=16)*8, v = 0..7

__device__ __forceinline__ v8f wmma3(v8f acc, const BfTile& ah, const BfTile& al,
                                     const BfTile& bh, const BfTile& bl) {
  acc = __builtin_amdgcn_wmma_f32_16x16x32_bf16(false, ah.v, false, bh.v,
                                                (short)0, acc, false, false);
  acc = __builtin_amdgcn_wmma_f32_16x16x32_bf16(false, ah.v, false, bl.v,
                                                (short)0, acc, false, false);
  acc = __builtin_amdgcn_wmma_f32_16x16x32_bf16(false, al.v, false, bh.v,
                                                (short)0, acc, false, false);
  return acc;
}

__device__ __forceinline__ v8f gemm_pair(v8f acc,
                                         const u16* __restrict__ Ahi,
                                         const u16* __restrict__ Alo,
                                         const u16* __restrict__ Bhi,
                                         const u16* __restrict__ Blo,
                                         int K, int row, int col,
                                         int akb, int bkb) {
  const u16* aH = Ahi + (size_t)row * K + akb;
  const u16* aL = Alo + (size_t)row * K + akb;
  const u16* bH = Bhi + (size_t)col * K + bkb;
  const u16* bL = Blo + (size_t)col * K + bkb;
  for (int k0 = 0; k0 < K; k0 += 32) {
    BfTile ah, al, bh, bl;
    ah.q[0] = *(const uint4*)(aH + k0);
    ah.q[1] = *(const uint4*)(aH + k0 + 16);
    al.q[0] = *(const uint4*)(aL + k0);
    al.q[1] = *(const uint4*)(aL + k0 + 16);
    bh.q[0] = *(const uint4*)(bH + k0);
    bh.q[1] = *(const uint4*)(bH + k0 + 16);
    bl.q[0] = *(const uint4*)(bL + k0);
    bl.q[1] = *(const uint4*)(bL + k0 + 16);
    acc = wmma3(acc, ah, al, bh, bl);
  }
  return acc;
}

// mode 0: out = dropout(relu(acc + bias)) (hop 1, deterministic hash mask)
// mode 1: out = acc + bias, cols < Nout only (hop 2 logits)
__global__ __launch_bounds__(256) void sage_gemm_wmma(
    const u16* __restrict__ A1hi, const u16* __restrict__ A1lo,
    const u16* __restrict__ B1hi, const u16* __restrict__ B1lo, int K1,
    const u16* __restrict__ A2hi, const u16* __restrict__ A2lo,
    const u16* __restrict__ B2hi, const u16* __restrict__ B2lo, int K2,
    const float* __restrict__ bias, float* __restrict__ out,
    int M, int Ntiles, int Nout, int ldout, int mode) {
  const int wid = blockIdx.x * (blockDim.x >> 5) + (threadIdx.x >> 5);
  const int nT = (M >> 4) * Ntiles;
  if (wid >= nT) return;                 // wave-uniform: EXEC all-ones in WMMA
  const int tm = wid / Ntiles;
  const int tn = wid - tm * Ntiles;
  const int lane = threadIdx.x & 31;
  const int hiHalf = lane >> 4;
  const int r = lane & 15;
  const int akb = hiHalf * 8;
  const int bkb = hiHalf * 8;
  const int row = tm * 16 + r;           // A row this lane feeds
  const int col = tn * 16 + r;           // B/C column this lane owns

  v8f acc = {};
  acc = gemm_pair(acc, A1hi, A1lo, B1hi, B1lo, K1, row, col, akb, bkb);
  acc = gemm_pair(acc, A2hi, A2lo, B2hi, B2lo, K2, row, col, akb, bkb);

  if (col < Nout) {
    const float bv = bias[col];
    const int mbase = tm * 16 + hiHalf * 8;
#pragma unroll
    for (int v = 0; v < 8; ++v) {
      const int m = mbase + v;
      float val = acc[v] + bv;
      if (mode == 0) {
        val = fmaxf(val, 0.0f);
        unsigned hsh = hash32((unsigned)(m * ldout + col) * 0x9E3779B9u + 42u);
        val = (hsh & 1u) ? val * 2.0f : 0.0f;   // p = 0.5, scale 1/(1-p)
      }
      out[(size_t)m * ldout + col] = val;
    }
  }
}

// One wave32 per row; cols <= 64.
__global__ void log_softmax_rows(float* __restrict__ o, int cols) {
  const int lane = threadIdx.x;
  float* p = o + (size_t)blockIdx.x * cols;
  float v0 = (lane < cols) ? p[lane] : -3.402823466e38f;
  float v1 = (lane + 32 < cols) ? p[lane + 32] : -3.402823466e38f;
  float m = fmaxf(v0, v1);
  for (int s = 16; s > 0; s >>= 1) m = fmaxf(m, __shfl_xor(m, s, 32));
  float e = 0.0f;
  if (lane < cols)      e += expf(v0 - m);
  if (lane + 32 < cols) e += expf(v1 - m);
  for (int s = 16; s > 0; s >>= 1) e += __shfl_xor(e, s, 32);
  const float ls = logf(e);
  if (lane < cols)      p[lane]      = v0 - m - ls;
  if (lane + 32 < cols) p[lane + 32] = v1 - m - ls;
}

// ------------------------------ host driver --------------------------------

static inline int blocksFor(long long n) {
  long long b = (n + 255) / 256;
  return (int)(b > 262144 ? 262144 : (b < 1 ? 1 : b));
}

extern "C" void kernel_launch(void* const* d_in, const int* in_sizes, int n_in,
                              void* d_out, int out_size, void* d_ws, size_t ws_size,
                              hipStream_t stream) {
  (void)n_in; (void)out_size; (void)ws_size;
  const float* x   = (const float*)d_in[0];
  const float* W1l = (const float*)d_in[1];
  const float* b1  = (const float*)d_in[2];
  const float* W1r = (const float*)d_in[3];
  const float* W2l = (const float*)d_in[4];
  const float* b2  = (const float*)d_in[5];
  const float* W2r = (const float*)d_in[6];
  const int* src1 = (const int*)d_in[7];   // harness: integer -> const int*
  const int* dst1 = (const int*)d_in[8];
  const int* src2 = (const int*)d_in[9];
  const int* dst2 = (const int*)d_in[10];
  const int E1 = in_sizes[7], E2 = in_sizes[9];

  const int ND1 = 20000, ND2 = 4000, IN = 512, HID = 256, OUT = 47, OUTP = 48;

  char* base = (char*)d_ws;
  size_t off = 0;
  auto alloc = [&](size_t bytes) -> void* {
    void* p = base + off;
    off += (bytes + 255) & ~(size_t)255;
    return p;
  };
  const long long nAgg1 = (long long)ND1 * IN;
  const long long nAgg2 = (long long)ND2 * HID;
  float* agg1  = (float*)alloc(nAgg1 * 4);
  float* cnt1  = (float*)alloc((size_t)ND1 * 4);
  float* agg2  = (float*)alloc(nAgg2 * 4);
  float* cnt2  = (float*)alloc((size_t)ND2 * 4);
  float* h     = (float*)alloc((size_t)ND1 * HID * 4);
  u16* A1hi  = (u16*)alloc(nAgg1 * 2);   u16* A1lo  = (u16*)alloc(nAgg1 * 2);
  u16* XDhi  = (u16*)alloc(nAgg1 * 2);   u16* XDlo  = (u16*)alloc(nAgg1 * 2);
  u16* A2hi  = (u16*)alloc(nAgg2 * 2);   u16* A2lo  = (u16*)alloc(nAgg2 * 2);
  u16* HDhi  = (u16*)alloc(nAgg2 * 2);   u16* HDlo  = (u16*)alloc(nAgg2 * 2);
  u16* W1lThi = (u16*)alloc((size_t)HID * IN * 2);
  u16* W1lTlo = (u16*)alloc((size_t)HID * IN * 2);
  u16* W1rThi = (u16*)alloc((size_t)HID * IN * 2);
  u16* W1rTlo = (u16*)alloc((size_t)HID * IN * 2);
  u16* W2lThi = (u16*)alloc((size_t)OUTP * HID * 2);
  u16* W2lTlo = (u16*)alloc((size_t)OUTP * HID * 2);
  u16* W2rThi = (u16*)alloc((size_t)OUTP * HID * 2);
  u16* W2rTlo = (u16*)alloc((size_t)OUTP * HID * 2);

  // hop 1: scatter-mean over 500K edges (HBM-bound, ~2 GB traffic)
  fz<<<blocksFor(nAgg1), 256, 0, stream>>>(agg1, nAgg1);
  fz<<<blocksFor(ND1),   256, 0, stream>>>(cnt1, ND1);
  fz<<<blocksFor(nAgg2), 256, 0, stream>>>(agg2, nAgg2);
  fz<<<blocksFor(ND2),   256, 0, stream>>>(cnt2, ND2);
  scatter_mean_accum<<<E1, 256, 0, stream>>>(src1, dst1, x, agg1, cnt1, IN);
  pack_mean_hilo<<<blocksFor(nAgg1), 256, 0, stream>>>(agg1, cnt1, A1hi, A1lo, ND1, IN);
  pack_hilo<<<blocksFor(nAgg1), 256, 0, stream>>>(x, XDhi, XDlo, nAgg1); // x[:ND1] contiguous
  pack_wT_hilo<<<blocksFor((long long)HID * IN), 256, 0, stream>>>(W1l, W1lThi, W1lTlo, IN, HID, HID);
  pack_wT_hilo<<<blocksFor((long long)HID * IN), 256, 0, stream>>>(W1r, W1rThi, W1rTlo, IN, HID, HID);

  // hop 1 GEMM: h = relu(agg1@W1l + b1 + x_dst@W1r) with dropout, via WMMA
  const int tiles1 = (ND1 / 16) * (HID / 16);
  sage_gemm_wmma<<<(tiles1 + 7) / 8, 256, 0, stream>>>(
      A1hi, A1lo, W1lThi, W1lTlo, IN,
      XDhi, XDlo, W1rThi, W1rTlo, IN,
      b1, h, ND1, HID / 16, HID, HID, 0);

  // hop 2
  pack_hilo<<<blocksFor(nAgg2), 256, 0, stream>>>(h, HDhi, HDlo, nAgg2); // h[:ND2]
  scatter_mean_accum<<<E2, 256, 0, stream>>>(src2, dst2, h, agg2, cnt2, HID);
  pack_mean_hilo<<<blocksFor(nAgg2), 256, 0, stream>>>(agg2, cnt2, A2hi, A2lo, ND2, HID);
  pack_wT_hilo<<<blocksFor((long long)OUTP * HID), 256, 0, stream>>>(W2l, W2lThi, W2lTlo, HID, OUT, OUTP);
  pack_wT_hilo<<<blocksFor((long long)OUTP * HID), 256, 0, stream>>>(W2r, W2rThi, W2rTlo, HID, OUT, OUTP);

  const int tiles2 = (ND2 / 16) * (OUTP / 16);
  sage_gemm_wmma<<<(tiles2 + 7) / 8, 256, 0, stream>>>(
      A2hi, A2lo, W2lThi, W2lTlo, HID,
      HDhi, HDlo, W2rThi, W2rTlo, HID,
      b2, (float*)d_out, ND2, OUTP / 16, OUT, OUT, 1);

  log_softmax_rows<<<ND2, 32, 0, stream>>>((float*)d_out, OUT);
}
